// Attention_3624952398027
// MI455X (gfx1250) — matrix operand order repaired and checked by reading, not compiled, not run on hardware
//
#include <hip/hip_runtime.h>
#include <hip/hip_bf16.h>

// Attention: out = softmax((xWq+bq)(xWk+bk)^T / sqrt(128)) @ (xWv+bv)
// Outputs concatenated: out (B*T*128 f32) then attn (B*T*T f32).
//
// MI455X plan: bf16 WMMA (v_wmma_f32_16x16x32_bf16, f32 accum) for all three
// GEMMs; full 16x4096 score stripe resident in LDS (256KB < 320KB/WGP) so
// softmax is exact and attn (268MB) is written to HBM exactly once.
// HBM traffic ~292MB -> ~12.5us floor; bf16 WMMA keeps the 37.6 GFLOP of
// matrix work well under that.

static constexpr int NB  = 4;
static constexpr int NT  = 4096;
static constexpr int DIN = 256;
static constexpr int DM  = 128;

typedef __attribute__((ext_vector_type(16))) __bf16 v16bf;
typedef __attribute__((ext_vector_type(8)))  float  v8f;

union FragBF { v16bf v; int4 q[2]; unsigned short u[16]; };

__device__ __forceinline__ unsigned short f2bf(float x) {
  unsigned int u = __float_as_uint(x);
  u += 0x7FFFu + ((u >> 16) & 1u);   // round-to-nearest-even
  return (unsigned short)(u >> 16);
}

__device__ __forceinline__ v8f wmma_bf16(v16bf a, v16bf b, v8f c) {
  // (neg_a, A, neg_b, B, c_mod, C, reuse_a, reuse_b)
  return __builtin_amdgcn_wmma_f32_16x16x32_bf16(false, a, false, b, (short)0, c,
                                                 false, false);
}

// ---------------------------------------------------------------------------
// Kernel 1: q/k/v projections. grid = (M/128, 3). block = 256 (8 waves).
// Each block: 128 rows x 128 cols of one projection, K = 256.
// Writes q,k row-major bf16; v transposed (vT[b][n][t]) for the PV GEMM.
// ---------------------------------------------------------------------------
__global__ void qkv_proj_kernel(const float* __restrict__ x,
                                const float* __restrict__ Wq, const float* __restrict__ bq,
                                const float* __restrict__ Wk, const float* __restrict__ bk,
                                const float* __restrict__ Wv, const float* __restrict__ bv,
                                unsigned short* __restrict__ qb,
                                unsigned short* __restrict__ kbuf,
                                unsigned short* __restrict__ vT) {
  extern __shared__ char smem[];
  unsigned short* Wt = (unsigned short*)smem;            // [128][256] bf16, W transposed (n-major)
  unsigned short* Xs = (unsigned short*)smem + DM * DIN; // [128][256] bf16, x rows

  const int tid  = threadIdx.x;
  const int wave = tid >> 5;
  const int lane = tid & 31;
  const int proj = blockIdx.y;
  const float* W   = (proj == 0) ? Wq : (proj == 1) ? Wk : Wv;
  const float* bia = (proj == 0) ? bq : (proj == 1) ? bk : bv;
  const int m0 = blockIdx.x * 128;

  // W f32 [k=256][n=128] -> Wt bf16 [n][k] (transposed so B-frags are contiguous)
  for (int idx = tid; idx < DIN * DM; idx += blockDim.x) {
    int k = idx >> 7, n = idx & 127;
    Wt[n * DIN + k] = f2bf(W[idx]);
  }
  // x rows m0..m0+127 f32 -> bf16
  for (int idx = tid; idx < 128 * DIN; idx += blockDim.x) {
    int r = idx >> 8, c = idx & 255;
    Xs[r * DIN + c] = f2bf(x[(size_t)(m0 + r) * DIN + c]);
  }
  __syncthreads();

  const int mloc = wave * 16;
  const int am = lane & 15;      // A row / B col / D col
  const int ah = lane >> 4;      // lane-half
  v8f acc[8];
  v8f zero = {0.f, 0.f, 0.f, 0.f, 0.f, 0.f, 0.f, 0.f};
#pragma unroll
  for (int nt = 0; nt < 8; ++nt) acc[nt] = zero;

  for (int kk = 0; kk < DIN; kk += 32) {
    FragBF A;  // A 16x32: elems 0..7 -> K=kb..kb+7, elems 8..15 -> K=kb+16..kb+23, kb=(lane>>4)*8
    const unsigned short* ap = Xs + (mloc + am) * DIN + kk + ah * 8;
    A.q[0] = *(const int4*)(ap);
    A.q[1] = *(const int4*)(ap + 16);
#pragma unroll
    for (int nt = 0; nt < 8; ++nt) {
      FragBF Bf; // B 32x16: elems 0..7 -> K=kb..kb+7, elems 8..15 -> K=kb+16..kb+23, kb=(lane>>4)*8, N = lane&15
      const unsigned short* bp = Wt + (nt * 16 + am) * DIN + kk + ah * 8;
      Bf.q[0] = *(const int4*)(bp);
      Bf.q[1] = *(const int4*)(bp + 16);
      acc[nt] = wmma_bf16(A.v, Bf.v, acc[nt]);
    }
  }

#pragma unroll
  for (int nt = 0; nt < 8; ++nt) {
    int col = nt * 16 + am;
    float bias_v = bia[col];
#pragma unroll
    for (int r = 0; r < 8; ++r) {  // D: M = r + 8*(lane>>4), N = lane&15
      int row = m0 + mloc + r + 8 * ah;
      unsigned short val = f2bf(acc[nt][r] + bias_v);
      if (proj == 0) {
        qb[(size_t)row * DM + col] = val;
      } else if (proj == 1) {
        kbuf[(size_t)row * DM + col] = val;
      } else {
        int bb = row >> 12;           // row / 4096
        int t  = row & (NT - 1);
        vT[(((size_t)bb * DM + col) << 12) + t] = val;
      }
    }
  }
}

// ---------------------------------------------------------------------------
// Kernel 2: fused scores + softmax + PV. grid = (T/16, B). block = 256.
// LDS: S[16][4096] f32 (256KB) + Qs[16][128] bf16 + red[16][16] f32.
// ---------------------------------------------------------------------------
__global__ void attn_kernel(const unsigned short* __restrict__ qb,
                            const unsigned short* __restrict__ kbuf,
                            const unsigned short* __restrict__ vT,
                            float* __restrict__ outp,
                            float* __restrict__ attnp) {
  extern __shared__ char smem[];
  float*          S   = (float*)smem;                                  // [16][4096]
  unsigned short* Qs  = (unsigned short*)(smem + 16 * NT * 4);         // [16][128]
  float*          red = (float*)(smem + 16 * NT * 4 + 16 * DM * 2);    // [16][16]

  const int tid  = threadIdx.x;
  const int wave = tid >> 5;
  const int lane = tid & 31;
  const int bb   = blockIdx.y;
  const int t0   = blockIdx.x * 16;
  const int am = lane & 15;
  const int ah = lane >> 4;
  const float scale = 0.08838834764831845f;  // 1/sqrt(128)

  // q stripe (already bf16) -> LDS
  for (int idx = tid; idx < 16 * DM; idx += blockDim.x) {
    int r = idx >> 7, c = idx & 127;
    Qs[idx] = qb[(size_t)(bb * NT + t0 + r) * DM + c];
  }
  __syncthreads();

  // A fragments (q stripe, K=0..127) held in registers
  FragBF A[4];
#pragma unroll
  for (int j = 0; j < 4; ++j) {
    const unsigned short* ap = Qs + am * DM + j * 32 + ah * 8;
    A[j].q[0] = *(const int4*)(ap);
    A[j].q[1] = *(const int4*)(ap + 16);
  }

  // scores: each wave owns 32 of the 256 column tiles; k streamed from L2
  v8f zero = {0.f, 0.f, 0.f, 0.f, 0.f, 0.f, 0.f, 0.f};
  for (int nt = wave * 32; nt < wave * 32 + 32; ++nt) {
    int n0 = nt * 16;
    v8f acc = zero;
    // B[k][n] = K[n0+n][k]: elems 0..7 -> d = j*32 + (lane>>4)*8 + i, elems 8..15 -> +16
    const unsigned short* krow = kbuf + (size_t)(bb * NT + n0 + am) * DM + ah * 8;
#pragma unroll
    for (int j = 0; j < 4; ++j) {
      FragBF Bf;
      Bf.q[0] = *(const int4*)(krow + j * 32);
      Bf.q[1] = *(const int4*)(krow + j * 32 + 16);
      acc = wmma_bf16(A[j].v, Bf.v, acc);
    }
#pragma unroll
    for (int r = 0; r < 8; ++r)
      S[(r + 8 * ah) * NT + n0 + am] = acc[r] * scale;
  }
  __syncthreads();

  // softmax: 16 threads per row, 256 elems each
  const int row = tid >> 4, seg = tid & 15;
  float* Srow = S + row * NT;
  float mx = -3.402823466e38f;
  for (int c = seg * 256; c < seg * 256 + 256; ++c) mx = fmaxf(mx, Srow[c]);
  red[row * 16 + seg] = mx;
  __syncthreads();
  float rmx = -3.402823466e38f;
#pragma unroll
  for (int j = 0; j < 16; ++j) rmx = fmaxf(rmx, red[row * 16 + j]);
  __syncthreads();
  float sum = 0.f;
  for (int c = seg * 256; c < seg * 256 + 256; ++c) {
    float e = __expf(Srow[c] - rmx);
    Srow[c] = e;
    sum += e;
  }
  red[row * 16 + seg] = sum;
  __syncthreads();
  float rs = 0.f;
#pragma unroll
  for (int j = 0; j < 16; ++j) rs += red[row * 16 + j];
  float inv = 1.0f / rs;
  float* arow = attnp + (size_t)(bb * NT + t0 + row) * NT;
  for (int c = seg * 256; c < seg * 256 + 256; ++c) {
    float p = Srow[c] * inv;
    Srow[c] = p;      // keep normalized P in LDS for PV GEMM
    arow[c] = p;      // attn written to HBM exactly once
  }
  __syncthreads();

  // out = P(16x4096) @ v(4096x128): wave owns one 16-wide N tile, K loop 4096
  const int n0 = wave * 16;
  v8f acc = zero;
  const unsigned short* vrow = vT + (((size_t)bb * DM + n0 + am) << 12) + ah * 8;
  const float* prow = S + am * NT + ah * 8;
  for (int kk = 0; kk < NT; kk += 32) {
    FragBF Af;  // convert P f32 -> bf16 on the fly (A layout: 0..7 @ kb, 8..15 @ kb+16)
    const float* pr = prow + kk;
    float4 x0 = *(const float4*)(pr);
    float4 x1 = *(const float4*)(pr + 4);
    float4 x2 = *(const float4*)(pr + 16);
    float4 x3 = *(const float4*)(pr + 20);
    Af.u[0]  = f2bf(x0.x); Af.u[1]  = f2bf(x0.y); Af.u[2]  = f2bf(x0.z); Af.u[3]  = f2bf(x0.w);
    Af.u[4]  = f2bf(x1.x); Af.u[5]  = f2bf(x1.y); Af.u[6]  = f2bf(x1.z); Af.u[7]  = f2bf(x1.w);
    Af.u[8]  = f2bf(x2.x); Af.u[9]  = f2bf(x2.y); Af.u[10] = f2bf(x2.z); Af.u[11] = f2bf(x2.w);
    Af.u[12] = f2bf(x3.x); Af.u[13] = f2bf(x3.y); Af.u[14] = f2bf(x3.z); Af.u[15] = f2bf(x3.w);
    FragBF Bf;  // B from vT: contiguous K
    Bf.q[0] = *(const int4*)(vrow + kk);
    Bf.q[1] = *(const int4*)(vrow + kk + 16);
    acc = wmma_bf16(Af.v, Bf.v, acc);
  }
#pragma unroll
  for (int r = 0; r < 8; ++r)
    outp[(size_t)(bb * NT + t0 + r + 8 * ah) * DM + n0 + am] = acc[r];
}

// ---------------------------------------------------------------------------
extern "C" void kernel_launch(void* const* d_in, const int* in_sizes, int n_in,
                              void* d_out, int out_size, void* d_ws, size_t ws_size,
                              hipStream_t stream) {
  const float* x  = (const float*)d_in[0];
  const float* Wq = (const float*)d_in[1];
  const float* bq = (const float*)d_in[2];
  const float* Wk = (const float*)d_in[3];
  const float* bk = (const float*)d_in[4];
  const float* Wv = (const float*)d_in[5];
  const float* bv = (const float*)d_in[6];

  float* outp  = (float*)d_out;                     // B*T*DM f32
  float* attnp = outp + (size_t)NB * NT * DM;       // B*T*T f32

  // workspace: q bf16, k bf16 (row-major), vT bf16 ([b][n][t]) -> 12 MB total
  unsigned short* qb   = (unsigned short*)d_ws;
  unsigned short* kbuf = qb + (size_t)NB * NT * DM;
  unsigned short* vT   = kbuf + (size_t)NB * NT * DM;

  // Kernel 1: 128 row-blocks x 3 projections
  {
    dim3 grid((NB * NT) / 128, 3, 1);
    size_t lds = (size_t)2 * DM * DIN * sizeof(unsigned short);  // 128 KB
    qkv_proj_kernel<<<grid, 256, lds, stream>>>(x, Wq, bq, Wk, bk, Wv, bv,
                                                qb, kbuf, vT);
  }

  // Kernel 2: one block per (batch, 16-query stripe)
  {
    dim3 grid(NT / 16, NB, 1);
    size_t lds = (size_t)16 * NT * 4 + (size_t)16 * DM * 2 + 16 * 16 * 4;  // ~261 KB
    attn_kernel<<<grid, 256, lds, stream>>>(qb, kbuf, vT, outp, attnp);
  }
}
